// ESN_4647154614366
// MI455X (gfx1250) — matrix operand order repaired and checked by reading, not compiled, not run on hardware
//
#include <hip/hip_runtime.h>
#include <hip/hip_bf16.h>
#include <math.h>

// ---------------------------------------------------------------------------
// ESN reservoir pipeline for MI455X (gfx1250), wave32, WMMA-based.
// Fixed problem shapes from the reference.
// ---------------------------------------------------------------------------
#define T_STEPS 4096
#define NRES    2048
#define DIN     64
#define KOUT    32
#define WASH    256
#define TW      (T_STEPS - WASH)   /* 3840 */
#define RIDGE   1e-6f
#define CG_ITERS 256
#define RES_WGS  32                /* reservoir workgroups (64 rows each) */

typedef __bf16 bf16_t;
typedef __attribute__((ext_vector_type(16))) __bf16 v16bf;
typedef __attribute__((ext_vector_type(8)))  __bf16 v8bf;
typedef __attribute__((ext_vector_type(8)))  float  v8f;
typedef __attribute__((ext_vector_type(2)))  float  v2f;
typedef __attribute__((ext_vector_type(4)))  unsigned v4u;
typedef __attribute__((ext_vector_type(8)))  int    v8i_;
typedef __attribute__((ext_vector_type(4)))  int    v4i_;

#ifndef __has_builtin
#define __has_builtin(x) 0
#endif
#if __has_builtin(__builtin_amdgcn_wmma_f32_16x16x4_f32)
#define HAVE_WMMA_F32X4 1
#else
#define HAVE_WMMA_F32X4 0
#endif
#if __has_builtin(__builtin_amdgcn_tensor_load_to_lds)
#define HAVE_TDM 1
#else
#define HAVE_TDM 0
#endif

// ---------------------------------------------------------------------------
// Grid-wide sense-reversing barrier (persistent kernels).
// ---------------------------------------------------------------------------
__device__ __forceinline__ void grid_sync(volatile unsigned* cnt,
                                          volatile unsigned* gen,
                                          unsigned nwg) {
  __syncthreads();
  if (threadIdx.x == 0) {
    __threadfence();
    unsigned g = *gen;
    if (atomicAdd((unsigned*)cnt, 1u) == nwg - 1u) {
      *cnt = 0u;
      __threadfence();
      atomicAdd((unsigned*)gen, 1u);
    } else {
      while (*gen == g) { __builtin_amdgcn_s_sleep(2); }
    }
    __threadfence();
  }
  __syncthreads();
}

// ---------------------------------------------------------------------------
// WMMA bf16 fragment loaders (ISA 7.12.2 layouts).
// A (row-major MxK, 16x32 tile): lane l<16: row=l,    K {k0..+7, k0+16..+23}
//                                lane l>=16: row=l-16, K {k0+8..+15, k0+24..+31}
// B as BT (NxK row-major):       lane l<16: col=l,    K {k0..+7, k0+16..+23}
//                                lane l>=16: col=l-16, K {k0+8..+15, k0+24..+31}
// All contiguous 16B loads.
// ---------------------------------------------------------------------------
__device__ __forceinline__ v16bf load_a_bf16(const bf16_t* A, int lda,
                                             int m0, int k0, int lane) {
  int row = m0 + (lane & 15);
  int kb  = k0 + ((lane >> 4) << 3);
  const bf16_t* p = A + (size_t)row * lda + kb;
  v8bf lo = *(const v8bf*)(p);
  v8bf hi = *(const v8bf*)(p + 16);
  v16bf a;
#pragma unroll
  for (int i = 0; i < 8; ++i) { a[i] = lo[i]; a[8 + i] = hi[i]; }
  return a;
}

__device__ __forceinline__ v16bf load_b_bf16(const bf16_t* BT, int ldb,
                                             int n0, int k0, int lane) {
  int row = n0 + (lane & 15);
  int kb  = k0 + ((lane >> 4) << 3);
  const bf16_t* p = BT + (size_t)row * ldb + kb;
  v8bf lo = *(const v8bf*)(p);
  v8bf hi = *(const v8bf*)(p + 16);
  v16bf b;
#pragma unroll
  for (int i = 0; i < 8; ++i) { b[i] = lo[i]; b[8 + i] = hi[i]; }
  return b;
}

// ---------------------------------------------------------------------------
// Generic NT GEMM: C(MxN, f32) = A(MxK bf16 row-major) * BT(NxK bf16)^T.
// Register-blocked MB x NB 16x16 tiles per wave: MB=2,NB=4 gives 8 WMMAs per
// 12 16B-loads (1.5 loads/WMMA). (M/16)%MB == 0, (N/16)%NB == 0, K%32 == 0.
// ---------------------------------------------------------------------------
template <int MB, int NB>
__global__ void gemm_nt_bf16(const bf16_t* __restrict__ A,
                             const bf16_t* __restrict__ BT,
                             float* __restrict__ C,
                             int M, int N, int K,
                             int lda, int ldb, int ldc) {
  const int lane = threadIdx.x & 31;
  const int wave = threadIdx.x >> 5;
  const int tiles_n = (N >> 4) / NB;
  const int tiles   = ((M >> 4) / MB) * tiles_n;
  const int tid = blockIdx.x * (blockDim.x >> 5) + wave;
  if (tid >= tiles) return;                 // wave-uniform: EXEC stays all-ones
  const int m0 = (tid / tiles_n) * (16 * MB);
  const int n0 = (tid % tiles_n) * (16 * NB);

  v8f c[MB][NB];
#pragma unroll
  for (int i = 0; i < MB; ++i)
#pragma unroll
    for (int j = 0; j < NB; ++j) c[i][j] = (v8f){};

  for (int k0 = 0; k0 < K; k0 += 32) {
    v16bf a[MB];
#pragma unroll
    for (int i = 0; i < MB; ++i)
      a[i] = load_a_bf16(A, lda, m0 + 16 * i, k0, lane);
    // warm the A/B streams ahead (lowers to global_prefetch_b8)
    __builtin_prefetch(A + (size_t)(m0 + (lane & 15)) * lda + k0 + 256, 0, 0);
    __builtin_prefetch(BT + (size_t)(n0 + (lane & 15)) * ldb + k0 + 256, 0, 0);
#pragma unroll
    for (int j = 0; j < NB; ++j) {
      v16bf b = load_b_bf16(BT, ldb, n0 + 16 * j, k0, lane);
#pragma unroll
      for (int i = 0; i < MB; ++i)
        c[i][j] = __builtin_amdgcn_wmma_f32_16x16x32_bf16(false, a[i], false, b,
                                                          (short)0, c[i][j],
                                                          false, false);
    }
  }
  const int coll = lane & 15;
#pragma unroll
  for (int i = 0; i < MB; ++i) {
    const int rb = m0 + 16 * i + ((lane >> 4) << 3);
#pragma unroll
    for (int j = 0; j < NB; ++j) {
      const int col = n0 + 16 * j + coll;
#pragma unroll
      for (int q = 0; q < 8; ++q)
        C[(size_t)(rb + q) * ldc + col] = c[i][j][q];
    }
  }
}

// ---------------------------------------------------------------------------
// Prep: bf16 conversions, y^T slice, zero state + barrier counters.
// ---------------------------------------------------------------------------
__global__ void prep_kernel(const float* __restrict__ W_res,
                            const float* __restrict__ W_in,
                            const float* __restrict__ u,
                            const float* __restrict__ y,
                            bf16_t* Wb, bf16_t* Wib, bf16_t* ub, bf16_t* ywTb,
                            bf16_t* xb0, bf16_t* xb1, unsigned* bars) {
  const size_t stride = (size_t)gridDim.x * blockDim.x;
  const size_t t0 = (size_t)blockIdx.x * blockDim.x + threadIdx.x;
  for (size_t i = t0; i < (size_t)NRES * NRES; i += stride) Wb[i]  = (bf16_t)W_res[i];
  for (size_t i = t0; i < (size_t)NRES * DIN;  i += stride) Wib[i] = (bf16_t)W_in[i];
  for (size_t i = t0; i < (size_t)T_STEPS * DIN; i += stride) ub[i] = (bf16_t)u[i];
  for (size_t i = t0; i < (size_t)KOUT * TW; i += stride) {
    int k = (int)(i / TW), t = (int)(i % TW);
    ywTb[i] = (bf16_t)y[(size_t)(WASH + t) * KOUT + k];
  }
  for (size_t i = t0; i < (size_t)NRES; i += stride) {
    xb0[i] = (bf16_t)0.f; xb1[i] = (bf16_t)0.f;
  }
  if (t0 < 8) bars[t0] = 0u;
}

// ---------------------------------------------------------------------------
// Persistent reservoir recurrence: x_{t+1} = tanh(W_res x_t + Wu_t).
// 32 WGs x 256 threads (8 waves). WG owns 64 rows; its W_res slice
// (64 x 2048 bf16 = 256 KB) sits in LDS (CDNA5: 320 KB/WGP), preloaded by the
// Tensor Data Mover when available (tensor_load_to_lds + s_wait_tensorcnt),
// else by a vector copy loop. Per-step matvec streams A-fragments from LDS.
// 8 waves = 4 row-tiles x 2 K-halves; within a wave, 2 interleaved WMMA
// accumulators break the serial D->C dependency (chain 32 -> 16).
// One grid barrier per step. Emits Xb (TWxN) and XbT (NxTW) bf16.
// ---------------------------------------------------------------------------
__global__ void __launch_bounds__(256, 1)
reservoir_kernel(const bf16_t* __restrict__ Wb, const float* __restrict__ Wu,
                 bf16_t* __restrict__ xb0, bf16_t* __restrict__ xb1,
                 bf16_t* __restrict__ Xb, bf16_t* __restrict__ XbT,
                 unsigned* bar_cnt, unsigned* bar_gen) {
  extern __shared__ char smem[];
  bf16_t* Wlds    = (bf16_t*)smem;                          // 64 x 2048 bf16
  float*  partial = (float*)(smem + (size_t)64 * NRES * 2); // 64 rows x 2 halves

  const int lane  = threadIdx.x & 31;
  const int wave  = threadIdx.x >> 5;
  const int tile  = wave >> 1;            // local row-tile 0..3
  const int khalf = wave & 1;             // K half 0..1
  const int row0  = blockIdx.x * 64;      // global row base of this WG
  const int m0l   = tile << 4;            // local row base
  const int kbase = khalf * (NRES / 2);

  // ---- one-time preload of this WG's W_res slice into LDS ----
#if HAVE_TDM
  if (wave == 0) {
    // D# for a 1D tile: 32768 x 8B units = 256 KB, contiguous.
    const unsigned long long gaddr =
        (unsigned long long)(const void*)(Wb + (size_t)row0 * NRES);
    const unsigned lds_base = __builtin_amdgcn_groupstaticsize(); // dyn LDS base
    v4u g0 = {0u, 0u, 0u, 0u};
    g0[0] = 1u;                                   // count=1 (user descriptor)
    g0[1] = lds_base;                             // lds_addr (bytes)
    g0[2] = (unsigned)(gaddr & 0xffffffffu);      // global_addr[31:0]
    g0[3] = (unsigned)((gaddr >> 32) & 0x1ffffffu) | (2u << 30); // [56:32]|type=2
    v8i_ g1 = {0, 0, 0, 0, 0, 0, 0, 0};
    g1[0] = (int)(3u << 16);        // workgroup_mask=0, data_size=3 (8B)
    g1[1] = (int)(32768u << 16);    // tensor_dim0[15:0]=32768 in bits[31:16]
    g1[2] = (int)(1u << 16);        // tensor_dim0[31:16]=0; tensor_dim1=1
    g1[3] = (int)(32768u << 16);    // tensor_dim1 hi=0; tile_dim0=32768
    g1[5] = 32768;                  // tensor_dim0_stride[31:0]
    v4i_ gz = {0, 0, 0, 0};
#if __clang_major__ >= 23
    v8i_ gz8 = {0, 0, 0, 0, 0, 0, 0, 0};
    __builtin_amdgcn_tensor_load_to_lds(g0, g1, gz, gz, gz8, 0);
#else
    __builtin_amdgcn_tensor_load_to_lds(g0, g1, gz, gz, 0);
#endif
#if __has_builtin(__builtin_amdgcn_s_wait_tensorcnt)
    __builtin_amdgcn_s_wait_tensorcnt(0);
#else
    asm volatile("s_wait_tensorcnt 0x0" ::: "memory");
#endif
  }
#else
  for (size_t i = (size_t)threadIdx.x * 8; i < (size_t)64 * NRES;
       i += (size_t)blockDim.x * 8) {
    *(v8bf*)(Wlds + i) = *(const v8bf*)(Wb + (size_t)row0 * NRES + i);
  }
#endif
  __syncthreads();

  const unsigned bmask = ((lane & 15) == 0) ? 0xffffffffu : 0u;

  for (int t = 0; t < T_STEPS; ++t) {
    const bf16_t* xin  = (t & 1) ? xb1 : xb0;
    bf16_t*       xout = (t & 1) ? xb0 : xb1;
    // two interleaved accumulators -> adjacent WMMAs are independent
    v8f c0 = {}, c1 = {};
    for (int k0 = kbase; k0 < kbase + NRES / 2; k0 += 64) {
#pragma unroll
      for (int half = 0; half < 2; ++half) {
        const int kc = k0 + 32 * half;
        // A fragment from LDS
        v16bf a;
        {
          int row = m0l + (lane & 15);
          int kb  = kc + ((lane >> 4) << 3);
          const bf16_t* p = Wlds + (size_t)row * NRES + kb;
          v8bf lo = *(const v8bf*)(p);
          v8bf hi = *(const v8bf*)(p + 16);
#pragma unroll
          for (int i = 0; i < 8; ++i) { a[i] = lo[i]; a[8 + i] = hi[i]; }
        }
        // B fragment: column 0 carries x. Branch-free: uniform 16B loads of
        // the x chunk in every lane, then dword-mask with a 0/-1 lane mask.
        v16bf b;
        {
          int kb = kc + ((lane >> 4) << 3);
          v4u lo = *(const v4u*)(xin + kb);
          v4u hi = *(const v4u*)(xin + kb + 16);
#pragma unroll
          for (int i = 0; i < 4; ++i) { lo[i] &= bmask; hi[i] &= bmask; }
          union { v4u u[2]; v16bf v; } cvt;
          cvt.u[0] = lo; cvt.u[1] = hi;
          b = cvt.v;
        }
        if (half == 0)
          c0 = __builtin_amdgcn_wmma_f32_16x16x32_bf16(false, a, false, b,
                                                       (short)0, c0,
                                                       false, false);
        else
          c1 = __builtin_amdgcn_wmma_f32_16x16x32_bf16(false, a, false, b,
                                                       (short)0, c1,
                                                       false, false);
      }
    }
    // lanes 0 and 16 hold column 0 of the D tiles: 8 partial row-sums each
    if ((lane & 15) == 0) {
      const int rloc = m0l + ((lane >> 4) << 3);
#pragma unroll
      for (int i = 0; i < 8; ++i)
        partial[(rloc + i) * 2 + khalf] = c0[i] + c1[i];
    }
    __syncthreads();
    // finalize: 64 rows -> threads 0..63
    if (threadIdx.x < 64) {
      const int r = row0 + threadIdx.x;
      float v = partial[threadIdx.x * 2 + 0] + partial[threadIdx.x * 2 + 1]
              + Wu[(size_t)t * NRES + r];
      v = tanhf(v);
      bf16_t bv = (bf16_t)v;
      xout[r] = bv;
      if (t >= WASH) {
        Xb[(size_t)(t - WASH) * NRES + r] = bv;
        XbT[(size_t)r * TW + (t - WASH)]  = bv;
      }
      if (t + 1 < T_STEPS)  // warm next step's Wu row (global_prefetch_b8)
        __builtin_prefetch(Wu + (size_t)(t + 1) * NRES + r, 0, 0);
    }
    grid_sync(bar_cnt, bar_gen, RES_WGS);
  }
}

// ---------------------------------------------------------------------------
// G += ridge * I ; W_out fp32 -> bf16.
// ---------------------------------------------------------------------------
__global__ void ridge_kernel(float* G) {
  int i = blockIdx.x * blockDim.x + threadIdx.x;
  if (i < NRES) G[(size_t)i * NRES + i] += RIDGE;
}
__global__ void cvt_wout_kernel(const float* __restrict__ XT, bf16_t* __restrict__ W) {
  int i = blockIdx.x * blockDim.x + threadIdx.x;
  if (i < KOUT * NRES) W[i] = (bf16_t)XT[i];
}

__device__ __forceinline__ float block_reduce(float v, float* red) {
  const int tid = threadIdx.x;
  red[tid] = v; __syncthreads();
  for (int s = 128; s > 0; s >>= 1) {
    if (tid < s) red[tid] += red[tid + s];
    __syncthreads();
  }
  float r = red[0]; __syncthreads();
  return r;
}

// ---------------------------------------------------------------------------
// Persistent batched CG on G (SPD) with 32 RHS, state transposed (32xN).
// WG g owns RHS column g -> all dots are intra-WG LDS reductions.
// Matvec APT = PT @ G (G symmetric): 256 wave-tiles of f32 WMMA 16x16x4,
// 4 interleaved accumulators (dependent chain 512 -> 128).
// 2 grid barriers per iteration; fixed CG_ITERS iterations.
// ---------------------------------------------------------------------------
__global__ void __launch_bounds__(256, 1)
cg_kernel(const float* __restrict__ G, const float* __restrict__ BrhsT,
          float* __restrict__ XT, float* __restrict__ RT,
          float* __restrict__ PT, float* __restrict__ APT,
          unsigned* bar_cnt, unsigned* bar_gen) {
  __shared__ float red[256];
  __shared__ float rr_s, pap_s, rn_s;
  const int tid  = threadIdx.x;
  const int lane = tid & 31;
  const int wave = tid >> 5;
  const int g    = blockIdx.x;            // owned RHS column (0..31)
  const int gw   = g * 8 + wave;          // global wave id = gemm tile id
  const int m0   = (gw >> 7) << 4;        // 0 or 16   (M = 32)
  const int n0   = (gw & 127) << 4;       // 0..2032   (N = 2048)

  // init: R = P = Brhs, X = 0, rr = <R,R> (own column only)
  float acc = 0.f;
  for (int r = tid; r < NRES; r += 256) {
    const size_t idx = (size_t)g * NRES + r;
    float b = BrhsT[idx];
    RT[idx] = b; PT[idx] = b; XT[idx] = 0.f;
    acc += b * b;
  }
  float rr0 = block_reduce(acc, red);
  if (tid == 0) rr_s = rr0;
  __syncthreads();
  grid_sync(bar_cnt, bar_gen, KOUT);

  for (int it = 0; it < CG_ITERS; ++it) {
    // ---- APT = PT @ G (NT form; fp32 WMMA 16x16x4, 4 accumulators) ----
    v8f cc[4];
#pragma unroll
    for (int uacc = 0; uacc < 4; ++uacc) cc[uacc] = (v8f){};
    const int arow = m0 + (lane & 15);
    const int brow = n0 + (lane & 15);
    for (int k0 = 0; k0 < NRES; k0 += 16) {
#pragma unroll
      for (int uacc = 0; uacc < 4; ++uacc) {
        const int kk = k0 + 4 * uacc + ((lane >> 4) << 1);
        v2f a = { PT[(size_t)arow * NRES + kk],
                  PT[(size_t)arow * NRES + kk + 1] };
        v2f b = { G[(size_t)brow * NRES + kk],
                  G[(size_t)brow * NRES + kk + 1] };
#if HAVE_WMMA_F32X4
        cc[uacc] = __builtin_amdgcn_wmma_f32_16x16x4_f32(false, a, false, b,
                                                         (short)0, cc[uacc],
                                                         false, false);
#else
#pragma unroll
        for (int i = 0; i < 8; ++i) cc[uacc][i] += a.x * b.x + a.y * b.y;
#endif
      }
    }
    {
      const int col = n0 + (lane & 15);
      const int rb  = m0 + ((lane >> 4) << 3);
#pragma unroll
      for (int i = 0; i < 8; ++i)
        APT[(size_t)(rb + i) * NRES + col] =
            (cc[0][i] + cc[1][i]) + (cc[2][i] + cc[3][i]);
    }
    grid_sync(bar_cnt, bar_gen, KOUT);

    // ---- column phase (own column g; WG-local reductions only) ----
    float pap = 0.f;
    for (int r = tid; r < NRES; r += 256) {
      const size_t idx = (size_t)g * NRES + r;
      pap += PT[idx] * APT[idx];
    }
    pap = block_reduce(pap, red);
    if (tid == 0) pap_s = pap;
    __syncthreads();
    const float alpha = rr_s / (pap_s + 1e-30f);

    float rn = 0.f;
    for (int r = tid; r < NRES; r += 256) {
      const size_t idx = (size_t)g * NRES + r;
      XT[idx] += alpha * PT[idx];
      float rv = RT[idx] - alpha * APT[idx];
      RT[idx] = rv;
      rn += rv * rv;
    }
    rn = block_reduce(rn, red);
    if (tid == 0) rn_s = rn;
    __syncthreads();
    const float beta = rn_s / (rr_s + 1e-30f);
    for (int r = tid; r < NRES; r += 256) {
      const size_t idx = (size_t)g * NRES + r;
      PT[idx] = RT[idx] + beta * PT[idx];
    }
    __syncthreads();
    if (tid == 0) rr_s = rn_s;
    grid_sync(bar_cnt, bar_gen, KOUT);
  }
}

// ---------------------------------------------------------------------------
// Host: workspace carve-out + fixed launch sequence (all on `stream`).
// ---------------------------------------------------------------------------
extern "C" void kernel_launch(void* const* d_in, const int* in_sizes, int n_in,
                              void* d_out, int out_size, void* d_ws, size_t ws_size,
                              hipStream_t stream) {
  (void)in_sizes; (void)n_in; (void)out_size; (void)ws_size;
  const float* u    = (const float*)d_in[0];
  const float* y    = (const float*)d_in[1];
  const float* Wres = (const float*)d_in[2];
  const float* Win  = (const float*)d_in[3];
  // d_in[4] = washout (==256, baked in as WASH)

  char* ws = (char*)d_ws;
  size_t off = 0;
  auto alloc = [&](size_t bytes) -> char* {
    char* p = ws + off;
    off = (off + bytes + 255) & ~(size_t)255;
    return p;
  };
  bf16_t* Wb    = (bf16_t*)alloc((size_t)NRES * NRES * 2);
  bf16_t* Wib   = (bf16_t*)alloc((size_t)NRES * DIN * 2);
  bf16_t* ub    = (bf16_t*)alloc((size_t)T_STEPS * DIN * 2);
  bf16_t* ywTb  = (bf16_t*)alloc((size_t)KOUT * TW * 2);
  float*  Wu    = (float*) alloc((size_t)T_STEPS * NRES * 4);
  bf16_t* xb0   = (bf16_t*)alloc((size_t)NRES * 2);
  bf16_t* xb1   = (bf16_t*)alloc((size_t)NRES * 2);
  bf16_t* Xb    = (bf16_t*)alloc((size_t)TW * NRES * 2);
  bf16_t* XbT   = (bf16_t*)alloc((size_t)NRES * TW * 2);
  float*  G     = (float*) alloc((size_t)NRES * NRES * 4);
  float*  BrhsT = (float*) alloc((size_t)KOUT * NRES * 4);
  float*  XT    = (float*) alloc((size_t)KOUT * NRES * 4);
  float*  RT    = (float*) alloc((size_t)KOUT * NRES * 4);
  float*  PT    = (float*) alloc((size_t)KOUT * NRES * 4);
  float*  APT   = (float*) alloc((size_t)KOUT * NRES * 4);
  bf16_t* WoutTb= (bf16_t*)alloc((size_t)KOUT * NRES * 2);
  unsigned* bars= (unsigned*)alloc(256);

  // dynamic LDS for reservoir: 64x2048 bf16 W-slice + 64x2 f32 partials
  const size_t res_lds = (size_t)64 * NRES * 2 + 64 * 2 * sizeof(float);

  // 1) conversions + zero state/barriers
  prep_kernel<<<2048, 256, 0, stream>>>(Wres, Win, u, y, Wb, Wib, ub, ywTb,
                                        xb0, xb1, bars);
  // 2) Wu = u @ W_in^T (NT: A=ub[TxD], BT=Wib[NxD]); tiles = (256/2)*(128/4)
  gemm_nt_bf16<2, 4><<<4096 / 8, 256, 0, stream>>>(ub, Wib, Wu,
                                                   T_STEPS, NRES, DIN,
                                                   DIN, DIN, NRES);
  // 3) sequential reservoir recurrence (persistent, LDS/TDM-resident W)
  reservoir_kernel<<<RES_WGS, 256, res_lds, stream>>>(Wb, Wu, xb0, xb1,
                                                      Xb, XbT,
                                                      bars + 0, bars + 1);
  // 4) G = Xw^T Xw (NT: A=XbT[NxTW], BT=XbT); tiles = (128/2)*(128/4) = 2048
  gemm_nt_bf16<2, 4><<<2048 / 8, 256, 0, stream>>>(XbT, XbT, G,
                                                   NRES, NRES, TW,
                                                   TW, TW, NRES);
  ridge_kernel<<<8, 256, 0, stream>>>(G);
  // 5) Brhs^T = yw^T Xw (NT: A=ywTb[KxTW], BT=XbT); tiles = (2/2)*(128/4)=32
  gemm_nt_bf16<2, 4><<<32 / 8, 256, 0, stream>>>(ywTb, XbT, BrhsT,
                                                 KOUT, NRES, TW,
                                                 TW, TW, NRES);
  // 6) solve G W_out = Brhs via batched CG (fp32 WMMA matvec)
  cg_kernel<<<KOUT, 256, 0, stream>>>(G, BrhsT, XT, RT, PT, APT,
                                      bars + 2, bars + 3);
  // 7) out = Xw @ W_out (NT: A=Xb[TWxN], BT=W_out^T bf16 [KxN]);
  //    tiles = (240/2)*(2/2) = 120
  cvt_wout_kernel<<<(KOUT * NRES) / 256, 256, 0, stream>>>(XT, WoutTb);
  gemm_nt_bf16<2, 2><<<120 / 8, 256, 0, stream>>>(Xb, WoutTb, (float*)d_out,
                                                  TW, KOUT, NRES,
                                                  NRES, NRES, KOUT);
}
